// DeltaCoupling_47571057770917
// MI455X (gfx1250) — matrix operand order repaired and checked by reading, not compiled, not run on hardware
//
#include <hip/hip_runtime.h>
#include <math.h>

typedef __attribute__((ext_vector_type(16))) _Float16 v16h;
typedef __attribute__((ext_vector_type(8)))  _Float16 v8h;
typedef __attribute__((ext_vector_type(8)))  float    v8f;

#define PI_OVER_CUT   2.6179938779914944f   // pi / 1.2
#define SILU_NORM_C   1.679177f
#define INV_SQRT50    0.1414213562373095f
#define INV_SQRT32    0.17677669529663687f
#define INV_SQRT3_C   0.5773502691896258f
#define HALF_INV_SQRT3 0.2886751345948129f
#define SQRT2_C       1.4142135623730951f   // sqrt(5)*sqrt(2/5)
#define SQRT6_5_C     1.0954451150103321f   // sqrt(3)*sqrt(2/5)
#define INV_SQRT128   0.08838834764831845f
#define INV_SQRT144   0.08333333333333333f
#define INV_SQRT160   0.07905694150420949f

// silu via v_rcp_f32 instead of the v_div_scale/v_div_fmas sequence
__device__ __forceinline__ float silu_f(float v) {
  return v * __builtin_amdgcn_rcpf(1.0f + __expf(-v));
}

// ---------------------------------------------------------------------------
// Stage fc weights -> f16, transposed [N][K] (padded K 50->64, N 41->48) so a
// lane's WMMA B-fragment (col fixed, 16 consecutive K) is one contiguous load.
// wst layout: W1t [32][64] @0, W2t [32][32] @2048, W3t [48][32] @3072
// ---------------------------------------------------------------------------
__global__ void stage_kernel(const float* __restrict__ fc0,
                             const float* __restrict__ fc1,
                             const float* __restrict__ fc2,
                             _Float16* __restrict__ wst) {
  int t = blockIdx.x * 256 + threadIdx.x;
  if (t < 2048) {
    int nn = t >> 6, k = t & 63;
    wst[t] = (_Float16)((k < 50) ? fc0[k * 32 + nn] : 0.0f);
  } else if (t < 3072) {
    int i = t - 2048; int nn = i >> 5, k = i & 31;
    wst[t] = (_Float16)fc1[k * 32 + nn];
  } else if (t < 4608) {
    int i = t - 3072; int nn = i >> 5, k = i & 31;
    wst[t] = (_Float16)((nn < 41) ? fc2[k * 41 + nn] : 0.0f);
  }
}

// ---------------------------------------------------------------------------
// Node init: x = concat(embed[Z], field); a = silu(silu(LN(raw@W1+b1))@W2+b2)
// ---------------------------------------------------------------------------
__global__ __launch_bounds__(256) void node_init(
    const int* __restrict__ Z, const float* __restrict__ field,
    const float* __restrict__ raw, const float* __restrict__ embed,
    const float* __restrict__ W1, const float* __restrict__ b1,
    const float* __restrict__ g, const float* __restrict__ b,
    const float* __restrict__ W2, const float* __restrict__ b2,
    float* __restrict__ x, float* __restrict__ a, int n) {
  int i = blockIdx.x * 256 + threadIdx.x;
  if (i >= n) return;
  int z = Z[i];
#pragma unroll
  for (int k = 0; k < 8; ++k) x[i * 16 + k] = embed[z * 8 + k];
#pragma unroll
  for (int k = 0; k < 8; ++k) x[i * 16 + 8 + k] = field[i * 8 + k];
  float r0 = raw[i * 3], r1 = raw[i * 3 + 1], r2 = raw[i * 3 + 2];
  float h[16];
#pragma unroll
  for (int k = 0; k < 16; ++k)
    h[k] = r0 * W1[k] + r1 * W1[16 + k] + r2 * W1[32 + k] + b1[k];
  float m = 0.f;
#pragma unroll
  for (int k = 0; k < 16; ++k) m += h[k];
  m *= 0.0625f;
  float v = 0.f;
#pragma unroll
  for (int k = 0; k < 16; ++k) { float d = h[k] - m; v += d * d; }
  v *= 0.0625f;
  float is = rsqrtf(v + 1e-5f);
  float hh[16];
#pragma unroll
  for (int k = 0; k < 16; ++k) {
    float t = (h[k] - m) * is * g[k] + b[k];
    hh[k] = silu_f(t);
  }
#pragma unroll
  for (int k = 0; k < 16; ++k) {
    float acc = b2[k];
    for (int j = 0; j < 16; ++j) acc += hh[j] * W2[j * 16 + k];
    a[i * 16 + k] = silu_f(acc);
  }
}

// ---------------------------------------------------------------------------
// Per-layer node kernel: sc and lin1 = fctp_scalar_node(x, a)
// ---------------------------------------------------------------------------
__global__ __launch_bounds__(256) void node_sclin1(
    const float* __restrict__ x, const float* __restrict__ a,
    const float* __restrict__ scW0, const float* __restrict__ scW1,
    const float* __restrict__ scW2, const float* __restrict__ l1W0,
    const float* __restrict__ l1W1, const float* __restrict__ l1W2,
    float* __restrict__ sc, float* __restrict__ lin1, int n) {
  __shared__ float sw[2048];
  for (int t = threadIdx.x; t < 1024; t += 256) { sw[t] = scW0[t]; sw[1024 + t] = l1W0[t]; }
  __syncthreads();
  int i = blockIdx.x * 256 + threadIdx.x;
  if (i >= n) return;
  float xv[16], av[16];
#pragma unroll
  for (int k = 0; k < 16; ++k) { xv[k] = x[i * 16 + k]; av[k] = a[i * 16 + k]; }
  float o0s[8] = {0}, o0l[8] = {0};
  for (int uu = 0; uu < 8; ++uu)
    for (int vv = 0; vv < 16; ++vv) {
      float t = xv[uu] * av[vv];
      const float* p1 = &sw[(uu * 16 + vv) * 8];
      const float* p2 = &sw[1024 + (uu * 16 + vv) * 8];
#pragma unroll
      for (int w = 0; w < 8; ++w) { o0s[w] += t * p1[w]; o0l[w] += t * p2[w]; }
    }
  float d1s = 0, d2s = 0, d1l = 0, d2l = 0;
#pragma unroll
  for (int vv = 0; vv < 16; ++vv) {
    d1s += av[vv] * scW1[vv]; d2s += av[vv] * scW2[vv];
    d1l += av[vv] * l1W1[vv]; d2l += av[vv] * l1W2[vv];
  }
  d1s *= 0.25f; d2s *= 0.25f; d1l *= 0.25f; d2l *= 0.25f;
#pragma unroll
  for (int w = 0; w < 8; ++w) {
    sc[i * 16 + w]   = o0s[w] * INV_SQRT128;
    lin1[i * 16 + w] = o0l[w] * INV_SQRT128;
  }
#pragma unroll
  for (int k = 0; k < 3; ++k) {
    sc[i * 16 + 8 + k]   = xv[8 + k] * d1s;
    lin1[i * 16 + 8 + k] = xv[8 + k] * d1l;
  }
#pragma unroll
  for (int k = 0; k < 5; ++k) {
    sc[i * 16 + 11 + k]   = xv[11 + k] * d2s;
    lin1[i * 16 + 11 + k] = xv[11 + k] * d2l;
  }
}

// ---------------------------------------------------------------------------
// Fused edge kernel (persistent waves): RBF -> 3-layer MLP via WMMA
// (16 edges/wave-tile) -> tensor-product messages -> atomic scatter into agg.
// B fragments for all 3 layers are hoisted into registers once per wave.
// ---------------------------------------------------------------------------
__global__ __launch_bounds__(256) void edge_kernel(
    const int* __restrict__ esrc, const int* __restrict__ edst,
    const float* __restrict__ edist, const float* __restrict__ esh,
    const float* __restrict__ lin1, const _Float16* __restrict__ wst,
    float* __restrict__ agg, int n_edges) {
  __shared__ _Float16 hbuf[8][16][32];  // per-wave 16x32 activation tile (f16)
  __shared__ float    wbuf[8][16][48];  // per-wave 16x48 final w tile (f32)
  const int lane = threadIdx.x & 31;
  const int wv   = threadIdx.x >> 5;
  const int ntiles = (n_edges + 15) >> 4;
  const int wave_id = blockIdx.x * 8 + wv;
  const int nwaves  = gridDim.x * 8;
  const int row  = lane & 15;   // edge row (A/M) and also B column
  const int half = lane >> 4;
  const int col  = row;
  const int khalf = half * 8;

  const _Float16* W1t = wst;          // [32][64]
  const _Float16* W2t = wst + 2048;   // [32][32]
  const _Float16* W3t = wst + 3072;   // [48][32]

  // ---- Hoist all B fragments (weights) into registers for the wave lifetime
  v16h B1[2][2], B2[2], B3[3];
#pragma unroll
  for (int nt = 0; nt < 2; ++nt) {
    const _Float16* wr = W1t + (nt * 16 + col) * 64;
    B1[nt][0] = *(const v16h*)(wr + half * 16);
    B1[nt][1] = *(const v16h*)(wr + 32 + half * 16);
    B2[nt] = *(const v16h*)(W2t + (nt * 16 + col) * 32 + half * 16);
  }
#pragma unroll
  for (int nt = 0; nt < 3; ++nt)
    B3[nt] = *(const v16h*)(W3t + (nt * 16 + col) * 32 + half * 16);

  for (int tile = wave_id; tile < ntiles; tile += nwaves) {
    int e = tile * 16 + row;
    bool valid = e < n_edges;
    if (!valid) e = n_edges - 1;

    // ---- RBF into A fragments (K padded 50 -> 64, two K=32 chunks)
    float d = edist[e];
    float cut = (d <= 1.2f) ? 0.5f * (__cosf(d * PI_OVER_CUT) + 1.0f) : 0.0f;
    v16h a0, a1;
#pragma unroll
    for (int i = 0; i < 16; ++i) {
      int kk = half * 16 + i;          // K in chunk 0 (0..31)
      float fd = d * (float)(kk + 1) * PI_OVER_CUT;
      float val = (fabsf(fd) < 1e-6f) ? 1.0f : __sinf(fd) * __builtin_amdgcn_rcpf(fd);
      a0[i] = (_Float16)(cut * val);
      int k1 = kk + 32;
      float r1 = 0.0f;
      if (k1 < 50) {
        float fd1 = d * (float)(k1 + 1) * PI_OVER_CUT;
        float v1 = (fabsf(fd1) < 1e-6f) ? 1.0f : __sinf(fd1) * __builtin_amdgcn_rcpf(fd1);
        r1 = cut * v1;
      }
      a1[i] = (_Float16)r1;
    }

    // ---- Layer 1: 64 -> 32 (two N tiles, two K chunks each)
#pragma unroll
    for (int nt = 0; nt < 2; ++nt) {
      v8f z = {};
      z = __builtin_amdgcn_wmma_f32_16x16x32_f16(false, a0, false, B1[nt][0], (short)0, z, false, false);
      z = __builtin_amdgcn_wmma_f32_16x16x32_f16(false, a1, false, B1[nt][1], (short)0, z, false, false);
#pragma unroll
      for (int r = 0; r < 8; ++r) {
        float v = z[r] * INV_SQRT50;
        hbuf[wv][khalf + r][nt * 16 + col] = (_Float16)(silu_f(v) * SILU_NORM_C);
      }
    }
    __asm__ volatile("" ::: "memory");   // DS in-order per wave; stop reordering

    // ---- Layer 2: 32 -> 32
    v16h a2;
    {
      const _Float16* hp = &hbuf[wv][row][0];
      v8h p0 = *(const v8h*)(hp + half * 16);
      v8h p1 = *(const v8h*)(hp + half * 16 + 8);
#pragma unroll
      for (int i = 0; i < 8; ++i) { a2[i] = p0[i]; a2[8 + i] = p1[i]; }
    }
    __asm__ volatile("" ::: "memory");
#pragma unroll
    for (int nt = 0; nt < 2; ++nt) {
      v8f z = {};
      z = __builtin_amdgcn_wmma_f32_16x16x32_f16(false, a2, false, B2[nt], (short)0, z, false, false);
#pragma unroll
      for (int r = 0; r < 8; ++r) {
        float v = z[r] * INV_SQRT32;
        hbuf[wv][khalf + r][nt * 16 + col] = (_Float16)(silu_f(v) * SILU_NORM_C);
      }
    }
    __asm__ volatile("" ::: "memory");

    // ---- Layer 3: 32 -> 48 (41 valid)
    v16h a3;
    {
      const _Float16* hp = &hbuf[wv][row][0];
      v8h p0 = *(const v8h*)(hp + half * 16);
      v8h p1 = *(const v8h*)(hp + half * 16 + 8);
#pragma unroll
      for (int i = 0; i < 8; ++i) { a3[i] = p0[i]; a3[8 + i] = p1[i]; }
    }
    __asm__ volatile("" ::: "memory");
#pragma unroll
    for (int nt = 0; nt < 3; ++nt) {
      v8f z = {};
      z = __builtin_amdgcn_wmma_f32_16x16x32_f16(false, a3, false, B3[nt], (short)0, z, false, false);
#pragma unroll
      for (int r = 0; r < 8; ++r)
        wbuf[wv][khalf + r][nt * 16 + col] = z[r] * INV_SQRT32;
    }
    __asm__ volatile("" ::: "memory");

    // ---- Tensor-product messages (lane pair l / l+16 share edge, split atomics)
    const float* wr = &wbuf[wv][row][0];   // w[0..40]
    int src = esrc[e], dst = edst[e];
    float s0a = esh[e * 6 + 0], s0b = esh[e * 6 + 1], s0c = esh[e * 6 + 2];
    float s1a = esh[e * 6 + 3], s1b = esh[e * 6 + 4], s1c = esh[e * 6 + 5];
    const float* gp = lin1 + (size_t)src * 16;
    float g0[8], g1[3], g2[5];
#pragma unroll
    for (int uu = 0; uu < 8; ++uu) g0[uu] = gp[uu];
    g1[0] = gp[8]; g1[1] = gp[9]; g1[2] = gp[10];
#pragma unroll
    for (int mm = 0; mm < 5; ++mm) g2[mm] = gp[11 + mm];

    float msg[49];
#pragma unroll
    for (int uu = 0; uu < 8; ++uu) {
      float dotv = wr[uu * 3] * s0a + wr[uu * 3 + 1] * s0b + wr[uu * 3 + 2] * s0c;
      msg[uu] = INV_SQRT3_C * g0[uu] * dotv;
      float wg = wr[24 + uu] * g0[uu];
      msg[8 + uu * 3 + 0] = wg * s1a;
      msg[8 + uu * 3 + 1] = wg * s1b;
      msg[8 + uu * 3 + 2] = wg * s1c;
    }
    {
      float d3a = wr[32] * s0a + wr[33] * s0b + wr[34] * s0c;
      msg[32] = INV_SQRT3_C * g1[0] * d3a;
      msg[33] = INV_SQRT3_C * g1[1] * d3a;
      msg[34] = INV_SQRT3_C * g1[2] * d3a;
      float g1s1 = g1[0] * s1a + g1[1] * s1b + g1[2] * s1c;
      msg[35] = INV_SQRT3_C * wr[35] * g1s1;
      float q = SQRT2_C * wr[36];
      msg[36] = q * 0.5f * (g1[0] * s1b + g1[1] * s1a);
      msg[37] = q * 0.5f * (g1[1] * s1c + g1[2] * s1b);
      msg[38] = q * (INV_SQRT3_C * g1[2] * s1c - HALF_INV_SQRT3 * (g1[0] * s1a + g1[1] * s1b));
      msg[39] = q * 0.5f * (g1[0] * s1c + g1[2] * s1a);
      msg[40] = q * 0.5f * (g1[0] * s1a - g1[1] * s1b);
      float d3b = wr[37] * s0a + wr[38] * s0b + wr[39] * s0c;
#pragma unroll
      for (int mm = 0; mm < 5; ++mm) msg[41 + mm] = INV_SQRT3_C * g2[mm] * d3b;
      float r6 = SQRT6_5_C * wr[40];
      msg[46] = r6 * (-HALF_INV_SQRT3 * g2[2] * s1a + 0.5f * g2[4] * s1a + 0.5f * g2[0] * s1b + 0.5f * g2[3] * s1c);
      msg[47] = r6 * (0.5f * g2[0] * s1a - HALF_INV_SQRT3 * g2[2] * s1b - 0.5f * g2[4] * s1b + 0.5f * g2[1] * s1c);
      msg[48] = r6 * (0.5f * g2[3] * s1a + 0.5f * g2[1] * s1b + INV_SQRT3_C * g2[2] * s1c);
    }
    if (valid) {
      float* basep = agg + (size_t)dst * 49;
#pragma unroll
      for (int i = 0; i < 49; ++i)
        if ((i < 25) == (half == 0)) unsafeAtomicAdd(basep + i, msg[i]);
    }
    __asm__ volatile("" ::: "memory");  // keep wbuf reads ordered vs next iter
  }
}

// ---------------------------------------------------------------------------
// Per-layer node update: agg mean -> lin2 + alpha gate -> residual into x
// ---------------------------------------------------------------------------
__global__ __launch_bounds__(256) void node_update(
    const float* __restrict__ a, const float* __restrict__ sc,
    const float* __restrict__ agg, const float* __restrict__ cnt,
    const float* __restrict__ W0, const float* __restrict__ W1,
    const float* __restrict__ W2, const float* __restrict__ alphaW,
    float* __restrict__ x, int n) {
  __shared__ float sw0[1152];
  __shared__ float swa[144];
  __shared__ float sw1[160];
  __shared__ float sw2[32];
  for (int t = threadIdx.x; t < 1152; t += 256) sw0[t] = W0[t];
  if (threadIdx.x < 144) swa[threadIdx.x] = alphaW[threadIdx.x];
  if (threadIdx.x < 160) sw1[threadIdx.x] = W1[threadIdx.x];
  if (threadIdx.x < 32)  sw2[threadIdx.x] = W2[threadIdx.x];
  __syncthreads();
  int i = blockIdx.x * 256 + threadIdx.x;
  if (i >= n) return;
  float inv = __builtin_amdgcn_rcpf(fmaxf(cnt[i], 1.0f));
  float ag[49];
#pragma unroll
  for (int k = 0; k < 49; ++k) ag[k] = agg[(size_t)i * 49 + k] * inv;
  float av[16];
#pragma unroll
  for (int k = 0; k < 16; ++k) av[k] = a[i * 16 + k];
  float A0[9];
#pragma unroll
  for (int uu = 0; uu < 8; ++uu) A0[uu] = ag[uu];
  A0[8] = ag[35];
  float o0[8] = {0};
  float alac = 0.f;
  for (int uu = 0; uu < 9; ++uu) {
    float aw = 0.f;
#pragma unroll
    for (int vv = 0; vv < 16; ++vv) aw += av[vv] * swa[uu * 16 + vv];
    alac += A0[uu] * aw;
    for (int vv = 0; vv < 16; ++vv) {
      float t = A0[uu] * av[vv];
      const float* p = &sw0[(uu * 16 + vv) * 8];
#pragma unroll
      for (int w = 0; w < 8; ++w) o0[w] += t * p[w];
    }
  }
  float alpha = tanhf(alac * INV_SQRT144);
  float A1[10][3];
#pragma unroll
  for (int uu = 0; uu < 8; ++uu)
#pragma unroll
    for (int k = 0; k < 3; ++k) A1[uu][k] = ag[8 + uu * 3 + k];
#pragma unroll
  for (int k = 0; k < 3; ++k) { A1[8][k] = ag[32 + k]; A1[9][k] = ag[46 + k]; }
  float o1[3] = {0};
  for (int uu = 0; uu < 10; ++uu) {
    float aw = 0.f;
#pragma unroll
    for (int vv = 0; vv < 16; ++vv) aw += av[vv] * sw1[uu * 16 + vv];
#pragma unroll
    for (int k = 0; k < 3; ++k) o1[k] += A1[uu][k] * aw;
  }
  float A2[2][5];
#pragma unroll
  for (int mm = 0; mm < 5; ++mm) { A2[0][mm] = ag[36 + mm]; A2[1][mm] = ag[41 + mm]; }
  float o2[5] = {0};
  for (int uu = 0; uu < 2; ++uu) {
    float aw = 0.f;
#pragma unroll
    for (int vv = 0; vv < 16; ++vv) aw += av[vv] * sw2[uu * 16 + vv];
#pragma unroll
    for (int mm = 0; mm < 5; ++mm) o2[mm] += A2[uu][mm] * aw;
  }
  float l2[16];
#pragma unroll
  for (int w = 0; w < 8; ++w) l2[w] = o0[w] * INV_SQRT144;
#pragma unroll
  for (int k = 0; k < 3; ++k) l2[8 + k] = o1[k] * INV_SQRT160;
#pragma unroll
  for (int mm = 0; mm < 5; ++mm) l2[11 + mm] = o2[mm] * INV_SQRT32;
#pragma unroll
  for (int k = 0; k < 16; ++k) {
    float xo = x[i * 16 + k];
    x[i * 16 + k] = 0.8f * (sc[i * 16 + k] + alpha * l2[k]) + 0.2f * xo;
  }
}

__global__ void count_kernel(const int* __restrict__ edst, float* __restrict__ cnt, int n) {
  int i = blockIdx.x * 256 + threadIdx.x;
  if (i < n) unsafeAtomicAdd(&cnt[edst[i]], 1.0f);
}

__global__ void pool_kernel(const int* __restrict__ batch, const float* __restrict__ x,
                            float* __restrict__ pooled, float* __restrict__ gcnt, int n) {
  int i = blockIdx.x * 256 + threadIdx.x;
  if (i >= n) return;
  int b = batch[i];
#pragma unroll
  for (int k = 0; k < 16; ++k) unsafeAtomicAdd(&pooled[b * 16 + k], x[i * 16 + k]);
  unsafeAtomicAdd(&gcnt[b], 1.0f);
}

__global__ void final_kernel(const float* __restrict__ pooled, const float* __restrict__ gcnt,
                             const float* __restrict__ u, const float* __restrict__ W1,
                             const float* __restrict__ b1, const float* __restrict__ g,
                             const float* __restrict__ b, const float* __restrict__ W2,
                             const float* __restrict__ b2, float* __restrict__ out, int ng) {
  int i = threadIdx.x;
  if (i >= ng) return;
  float z[20];
  float inv = 1.0f / fmaxf(gcnt[i], 1.0f);
#pragma unroll
  for (int k = 0; k < 16; ++k) z[k] = pooled[i * 16 + k] * inv;
#pragma unroll
  for (int k = 0; k < 4; ++k) z[16 + k] = u[i * 4 + k];
  float t[32];
  for (int k = 0; k < 32; ++k) {
    float acc = b1[k];
#pragma unroll
    for (int j = 0; j < 20; ++j) acc += z[j] * W1[j * 32 + k];
    t[k] = acc;
  }
  float m = 0.f;
#pragma unroll
  for (int k = 0; k < 32; ++k) m += t[k];
  m *= (1.0f / 32.0f);
  float v = 0.f;
#pragma unroll
  for (int k = 0; k < 32; ++k) { float d = t[k] - m; v += d * d; }
  v *= (1.0f / 32.0f);
  float is = rsqrtf(v + 1e-5f);
  float o = b2[0];
#pragma unroll
  for (int k = 0; k < 32; ++k) {
    float s = (t[k] - m) * is * g[k] + b[k];
    o += silu_f(s) * W2[k];
  }
  out[i] = o;
}

// ---------------------------------------------------------------------------
extern "C" void kernel_launch(void* const* d_in, const int* in_sizes, int n_in,
                              void* d_out, int out_size, void* d_ws, size_t ws_size,
                              hipStream_t stream) {
  (void)n_in; (void)ws_size;
  const int*   Z     = (const int*)d_in[0];
  const float* field = (const float*)d_in[1];
  const float* raw   = (const float*)d_in[2];
  const int*   esrc  = (const int*)d_in[3];
  const int*   edst  = (const int*)d_in[4];
  const float* edist = (const float*)d_in[5];
  const float* esh   = (const float*)d_in[6];
  const int*   batch = (const int*)d_in[7];
  const float* uG    = (const float*)d_in[8];
  const float* embed = (const float*)d_in[9];
  const float* naW1 = (const float*)d_in[10];
  const float* nab1 = (const float*)d_in[11];
  const float* nag  = (const float*)d_in[12];
  const float* nab  = (const float*)d_in[13];
  const float* naW2 = (const float*)d_in[14];
  const float* nab2 = (const float*)d_in[15];
  const float* mW1 = (const float*)d_in[42];
  const float* mb1 = (const float*)d_in[43];
  const float* mg  = (const float*)d_in[44];
  const float* mb  = (const float*)d_in[45];
  const float* mW2 = (const float*)d_in[46];
  const float* mb2 = (const float*)d_in[47];

  int n_nodes  = in_sizes[0];
  int n_edges  = in_sizes[3];
  int n_graphs = out_size;

  char* wsb = (char*)d_ws;
  size_t off = 0;
  auto alloc = [&](size_t bytes) -> void* {
    void* p = wsb + off;
    off += (bytes + 255) & ~(size_t)255;
    return p;
  };
  float* x      = (float*)alloc((size_t)n_nodes * 16 * 4);
  float* a      = (float*)alloc((size_t)n_nodes * 16 * 4);
  float* sc     = (float*)alloc((size_t)n_nodes * 16 * 4);
  float* lin1   = (float*)alloc((size_t)n_nodes * 16 * 4);
  float* agg    = (float*)alloc((size_t)n_nodes * 49 * 4);
  float* cnt    = (float*)alloc((size_t)n_nodes * 4);
  float* pooled = (float*)alloc((size_t)n_graphs * 16 * 4);
  float* gcnt   = (float*)alloc((size_t)n_graphs * 4);
  _Float16* wst0 = (_Float16*)alloc(4608 * 2);
  _Float16* wst1 = (_Float16*)alloc(4608 * 2);
  _Float16* wsts[2] = {wst0, wst1};

  int nb = (n_nodes + 255) / 256;
  hipMemsetAsync(cnt, 0, (size_t)n_nodes * 4, stream);
  node_init<<<nb, 256, 0, stream>>>(Z, field, raw, embed, naW1, nab1, nag, nab,
                                    naW2, nab2, x, a, n_nodes);
  count_kernel<<<(n_edges + 255) / 256, 256, 0, stream>>>(edst, cnt, n_edges);
  for (int L = 0; L < 2; ++L) {
    int base = 16 + 13 * L;
    stage_kernel<<<18, 256, 0, stream>>>((const float*)d_in[base + 6],
                                         (const float*)d_in[base + 7],
                                         (const float*)d_in[base + 8], wsts[L]);
  }
  int ntiles = (n_edges + 15) / 16;
  int eblocks = (ntiles + 7) / 8;
  if (eblocks > 1024) eblocks = 1024;   // persistent waves, grid-stride over tiles
  for (int L = 0; L < 2; ++L) {
    int base = 16 + 13 * L;
    const float* scW0  = (const float*)d_in[base + 0];
    const float* scW1  = (const float*)d_in[base + 1];
    const float* scW2  = (const float*)d_in[base + 2];
    const float* l1W0  = (const float*)d_in[base + 3];
    const float* l1W1  = (const float*)d_in[base + 4];
    const float* l1W2  = (const float*)d_in[base + 5];
    const float* l2W0  = (const float*)d_in[base + 9];
    const float* l2W1  = (const float*)d_in[base + 10];
    const float* l2W2  = (const float*)d_in[base + 11];
    const float* alphW = (const float*)d_in[base + 12];
    node_sclin1<<<nb, 256, 0, stream>>>(x, a, scW0, scW1, scW2, l1W0, l1W1, l1W2,
                                        sc, lin1, n_nodes);
    hipMemsetAsync(agg, 0, (size_t)n_nodes * 49 * 4, stream);
    edge_kernel<<<eblocks, 256, 0, stream>>>(esrc, edst, edist, esh, lin1,
                                             wsts[L], agg, n_edges);
    node_update<<<nb, 256, 0, stream>>>(a, sc, agg, cnt, l2W0, l2W1, l2W2, alphW,
                                        x, n_nodes);
  }
  hipMemsetAsync(pooled, 0, (size_t)n_graphs * 16 * 4, stream);
  hipMemsetAsync(gcnt, 0, (size_t)n_graphs * 4, stream);
  pool_kernel<<<nb, 256, 0, stream>>>(batch, x, pooled, gcnt, n_nodes);
  final_kernel<<<1, 64, 0, stream>>>(pooled, gcnt, uG, mW1, mb1, mg, mb, mW2, mb2,
                                     (float*)d_out, n_graphs);
}
